// SparseStructuralGATLayer_88648124989883
// MI455X (gfx1250) — compile-verified, matrix operand order checked by reading, not run on hardware
//
#include <hip/hip_runtime.h>
#include <hip/hip_bf16.h>
#include <stdint.h>

#define DIM 128
#define NEG_SLOPE 0.2f
#define LN_EPS 1e-5f

typedef __attribute__((ext_vector_type(16))) _Float16 v16h;
typedef __attribute__((ext_vector_type(8)))  _Float16 v8h;
typedef __attribute__((ext_vector_type(4)))  _Float16 v4h;
typedef __attribute__((ext_vector_type(8)))  float    v8f;

// ---------------------------------------------------------------------------
// 0) init: Z (=d_out) to 0, segmax to -inf, denom to 0
// ---------------------------------------------------------------------------
__global__ __launch_bounds__(256) void gat_init(float* __restrict__ Z,
                                                float* __restrict__ segmax,
                                                float* __restrict__ denom,
                                                int n) {
    long long idx = (long long)blockIdx.x * 256 + threadIdx.x;
    if (idx < (long long)n * DIM) Z[idx] = 0.0f;
    if (idx < n) {
        segmax[idx] = -3.402823466e38f;
        denom[idx]  = 0.0f;
    }
}

// ---------------------------------------------------------------------------
// 1) convert W [128,128] f32 -> f16 (64KB -> 32KB; stays hot in L2)
// ---------------------------------------------------------------------------
__global__ __launch_bounds__(256) void gat_convW(const float* __restrict__ W,
                                                 _Float16* __restrict__ Wh) {
    int i = blockIdx.x * 256 + threadIdx.x;
    if (i < DIM * DIM) Wh[i] = (_Float16)W[i];
}

// ---------------------------------------------------------------------------
// 2) H = X @ W^T via v_wmma_f32_16x16x32_f16
//    One block = 32 rows of H; 8 waves cover the 8 16-col tiles.
//    Each wave holds one B fragment per K-step and feeds TWO row tiles
//    (two accumulators) -> 8 WMMAs per wave, 2x B-register reuse.
// ---------------------------------------------------------------------------
__global__ __launch_bounds__(256) void gat_gemm(const float* __restrict__ X,
                                                const _Float16* __restrict__ Wh,
                                                float* __restrict__ H, int n) {
    __shared__ __align__(16) _Float16 Ash[32][DIM + 8];  // row pitch 272B (16B mult)

    const int rowbase = blockIdx.x * 32;
    const bool full   = (rowbase + 32 <= n);

    // Stage the 32x128 X tile into LDS as f16 (float4 loads, 8B f16 stores)
    if (full) {
        for (int i = threadIdx.x; i < 32 * DIM / 4; i += 256) {
            int r  = i >> 5;           // 32 float4 per row
            int c4 = (i & 31) * 4;
            float4 v = *(const float4*)&X[(size_t)(rowbase + r) * DIM + c4];
            v4h h; h[0] = (_Float16)v.x; h[1] = (_Float16)v.y;
                   h[2] = (_Float16)v.z; h[3] = (_Float16)v.w;
            *(v4h*)&Ash[r][c4] = h;
        }
    } else {
        for (int i = threadIdx.x; i < 32 * DIM; i += 256) {
            int r = i >> 7, c = i & (DIM - 1);
            int gr = rowbase + r;
            Ash[r][c] = (gr < n) ? (_Float16)X[(size_t)gr * DIM + c] : (_Float16)0.0f;
        }
    }
    __syncthreads();

    const int wave = threadIdx.x >> 5;     // 0..7 -> output col tile
    const int lane = threadIdx.x & 31;
    const int m    = lane & 15;            // row (A) / col (B) within tile
    const int kb   = (lane < 16) ? 0 : 8;  // K sub-block per half-wave
    const int j    = wave * 16 + m;        // row of W == output column of H

    v8f acc0 = {}, acc1 = {};
#pragma unroll
    for (int kk = 0; kk < DIM; kk += 32) {
        // B fragment (shared by both row tiles)
        v8h b0 = *(const v8h*)&Wh[(size_t)j * DIM + kk + kb];
        v8h b1 = *(const v8h*)&Wh[(size_t)j * DIM + kk + kb + 16];
        v16h b;
#pragma unroll
        for (int i = 0; i < 8; ++i) { b[i] = b0[i]; b[i + 8] = b1[i]; }

        // A fragments for row tile 0 (rows 0-15) and tile 1 (rows 16-31)
        v8h a00 = *(const v8h*)&Ash[m][kk + kb];
        v8h a01 = *(const v8h*)&Ash[m][kk + kb + 16];
        v8h a10 = *(const v8h*)&Ash[m + 16][kk + kb];
        v8h a11 = *(const v8h*)&Ash[m + 16][kk + kb + 16];
        v16h a0, a1;
#pragma unroll
        for (int i = 0; i < 8; ++i) {
            a0[i] = a00[i]; a0[i + 8] = a01[i];
            a1[i] = a10[i]; a1[i + 8] = a11[i];
        }
        acc0 = __builtin_amdgcn_wmma_f32_16x16x32_f16(false, a0, false, b,
                                                      (short)0, acc0, false, false);
        acc1 = __builtin_amdgcn_wmma_f32_16x16x32_f16(false, a1, false, b,
                                                      (short)0, acc1, false, false);
    }

    // C/D layout: VGPR r -> M=r (lanes 0-15) / M=r+8 (lanes 16-31); N = lane&15
    const int mb   = (lane < 16) ? 0 : 8;
    const int nOut = wave * 16 + (lane & 15);
    if (full) {
#pragma unroll
        for (int r = 0; r < 8; ++r)
            H[(size_t)(rowbase + mb + r) * DIM + nOut] = acc0[r];
#pragma unroll
        for (int r = 0; r < 8; ++r)
            H[(size_t)(rowbase + 16 + mb + r) * DIM + nOut] = acc1[r];
    } else {
#pragma unroll
        for (int r = 0; r < 8; ++r) {
            int g0 = rowbase + mb + r;
            if (g0 < n) H[(size_t)g0 * DIM + nOut] = acc0[r];
            int g1 = rowbase + 16 + mb + r;
            if (g1 < n) H[(size_t)g1 * DIM + nOut] = acc1[r];
        }
    }
}

// ---------------------------------------------------------------------------
// 3) e_src = H @ a_src, e_dst = H @ a_dst  (one wave32 per node)
// ---------------------------------------------------------------------------
__global__ __launch_bounds__(256) void gat_dots(const float* __restrict__ H,
                                                const float* __restrict__ a_src,
                                                const float* __restrict__ a_dst,
                                                float* __restrict__ es,
                                                float* __restrict__ ed, int n) {
    int node = blockIdx.x * 8 + (threadIdx.x >> 5);
    int lane = threadIdx.x & 31;
    if (node >= n) return;
    float4 h  = ((const float4*)(H + (size_t)node * DIM))[lane];
    float4 as = ((const float4*)a_src)[lane];
    float4 ad = ((const float4*)a_dst)[lane];
    float s = h.x * as.x + h.y * as.y + h.z * as.z + h.w * as.w;
    float d = h.x * ad.x + h.y * ad.y + h.z * ad.z + h.w * ad.w;
#pragma unroll
    for (int off = 16; off > 0; off >>= 1) {
        s += __shfl_down(s, off);
        d += __shfl_down(d, off);
    }
    if (lane == 0) { es[node] = s; ed[node] = d; }
}

// ---------------------------------------------------------------------------
// 4) per-edge logits + segment-max (CAS-based float atomic max)
// ---------------------------------------------------------------------------
__device__ inline void atomicMaxF(float* addr, float val) {
    unsigned int* ua = (unsigned int*)addr;
    unsigned int old = *ua;
    while (__uint_as_float(old) < val) {
        unsigned int assumed = old;
        old = atomicCAS(ua, assumed, __float_as_uint(val));
        if (old == assumed) break;
    }
}

__global__ __launch_bounds__(256) void gat_logits(const int* __restrict__ src,
                                                  const int* __restrict__ dst,
                                                  const float* __restrict__ Mw,
                                                  const float* __restrict__ es,
                                                  const float* __restrict__ ed,
                                                  float* __restrict__ logits,
                                                  float* __restrict__ segmax, int E) {
    int e = blockIdx.x * 256 + threadIdx.x;
    if (e >= E) return;
    int s = src[e], d = dst[e];
    float x  = es[s] + ed[d];
    float lr = (x > 0.0f) ? x : NEG_SLOPE * x;
    float lg = Mw[e] * lr;
    logits[e] = lg;
    atomicMaxF(&segmax[s], lg);
}

// ---------------------------------------------------------------------------
// 5) ex = exp(logit - segmax[src]); denom[src] += ex
// ---------------------------------------------------------------------------
__global__ __launch_bounds__(256) void gat_expsum(const int* __restrict__ src,
                                                  const float* __restrict__ logits,
                                                  const float* __restrict__ segmax,
                                                  float* __restrict__ exv,
                                                  float* __restrict__ denom, int E) {
    int e = blockIdx.x * 256 + threadIdx.x;
    if (e >= E) return;
    int s = src[e];
    float ex = __expf(logits[e] - segmax[s]);
    exv[e] = ex;
    atomicAdd(&denom[s], ex);
}

// ---------------------------------------------------------------------------
// 6) Z[src] += alpha * H[dst]  (one wave32 per edge)
//    Lane 0 resolves edge metadata; wave32 shuffles broadcast it.
// ---------------------------------------------------------------------------
__global__ __launch_bounds__(256) void gat_scatter(const int* __restrict__ src,
                                                   const int* __restrict__ dst,
                                                   const float* __restrict__ exv,
                                                   const float* __restrict__ denom,
                                                   const float* __restrict__ H,
                                                   float* __restrict__ Z, int E) {
    int e = blockIdx.x * 8 + (threadIdx.x >> 5);
    int lane = threadIdx.x & 31;
    if (e >= E) return;
    int s = 0, d = 0;
    float alpha = 0.0f;
    if (lane == 0) {
        s = src[e];
        d = dst[e];
        alpha = exv[e] / denom[s];
    }
    s = __shfl(s, 0);
    d = __shfl(d, 0);
    alpha = __shfl(alpha, 0);

    float4 h = ((const float4*)(H + (size_t)d * DIM))[lane];
    float* zp = Z + (size_t)s * DIM + lane * 4;
    atomicAdd(zp + 0, h.x * alpha);
    atomicAdd(zp + 1, h.y * alpha);
    atomicAdd(zp + 2, h.z * alpha);
    atomicAdd(zp + 3, h.w * alpha);
}

// ---------------------------------------------------------------------------
// 7) finalize: isolated-node fallback, residual, LayerNorm, ELU (in place on Z)
// ---------------------------------------------------------------------------
__global__ __launch_bounds__(256) void gat_finalize(float* __restrict__ Z,
                                                    const float* __restrict__ H,
                                                    const float* __restrict__ X,
                                                    const float* __restrict__ denom,
                                                    const float* __restrict__ gamma,
                                                    const float* __restrict__ beta,
                                                    int n) {
    int node = blockIdx.x * 8 + (threadIdx.x >> 5);
    int lane = threadIdx.x & 31;
    if (node >= n) return;

    float4 z = ((const float4*)(Z + (size_t)node * DIM))[lane];
    if (denom[node] <= 0.0f) {              // isolated node: Z = H
        z = ((const float4*)(H + (size_t)node * DIM))[lane];
    }
    float4 x = ((const float4*)(X + (size_t)node * DIM))[lane];
    z.x += x.x; z.y += x.y; z.z += x.z; z.w += x.w;   // residual

    float s = z.x + z.y + z.z + z.w;
#pragma unroll
    for (int off = 16; off > 0; off >>= 1) s += __shfl_xor(s, off);
    float mu = s * (1.0f / DIM);

    float dx0 = z.x - mu, dx1 = z.y - mu, dx2 = z.z - mu, dx3 = z.w - mu;
    float v = dx0 * dx0 + dx1 * dx1 + dx2 * dx2 + dx3 * dx3;
#pragma unroll
    for (int off = 16; off > 0; off >>= 1) v += __shfl_xor(v, off);
    float rs = rsqrtf(v * (1.0f / DIM) + LN_EPS);

    float4 g = ((const float4*)gamma)[lane];
    float4 b = ((const float4*)beta)[lane];
    float4 o;
    o.x = dx0 * rs * g.x + b.x;
    o.y = dx1 * rs * g.y + b.y;
    o.z = dx2 * rs * g.z + b.z;
    o.w = dx3 * rs * g.w + b.w;
    // ELU
    o.x = (o.x > 0.0f) ? o.x : (__expf(o.x) - 1.0f);
    o.y = (o.y > 0.0f) ? o.y : (__expf(o.y) - 1.0f);
    o.z = (o.z > 0.0f) ? o.z : (__expf(o.z) - 1.0f);
    o.w = (o.w > 0.0f) ? o.w : (__expf(o.w) - 1.0f);
    ((float4*)(Z + (size_t)node * DIM))[lane] = o;
}

// ---------------------------------------------------------------------------
extern "C" void kernel_launch(void* const* d_in, const int* in_sizes, int n_in,
                              void* d_out, int out_size, void* d_ws, size_t ws_size,
                              hipStream_t stream) {
    const float* X      = (const float*)d_in[0];
    const int*   ei     = (const int*)d_in[1];
    const float* Mw     = (const float*)d_in[2];
    const float* W      = (const float*)d_in[3];
    const float* a_src  = (const float*)d_in[4];
    const float* a_dst  = (const float*)d_in[5];
    const float* gamma  = (const float*)d_in[6];
    const float* beta   = (const float*)d_in[7];

    const int n = in_sizes[0] / DIM;
    const int E = in_sizes[2];
    const int* srcI = ei;
    const int* dstI = ei + E;

    // Workspace layout
    float* H      = (float*)d_ws;                    // n*128
    float* es     = H + (size_t)n * DIM;             // n
    float* ed     = es + n;                          // n
    float* segmax = ed + n;                          // n
    float* denom  = segmax + n;                      // n
    float* logits = denom + n;                       // E
    float* exv    = logits + E;                      // E
    uintptr_t p = (uintptr_t)(exv + E);
    p = (p + 15) & ~(uintptr_t)15;
    _Float16* Wh = (_Float16*)p;                     // 128*128 f16
    float* Z = (float*)d_out;                        // accumulate in d_out

    gat_init<<<((long long)n * DIM + 255) / 256, 256, 0, stream>>>(Z, segmax, denom, n);
    gat_convW<<<(DIM * DIM + 255) / 256, 256, 0, stream>>>(W, Wh);
    gat_gemm<<<(n + 31) / 32, 256, 0, stream>>>(X, Wh, H, n);
    gat_dots<<<(n + 7) / 8, 256, 0, stream>>>(H, a_src, a_dst, es, ed, n);
    gat_logits<<<(E + 255) / 256, 256, 0, stream>>>(srcI, dstI, Mw, es, ed, logits, segmax, E);
    gat_expsum<<<(E + 255) / 256, 256, 0, stream>>>(srcI, logits, segmax, exv, denom, E);
    gat_scatter<<<(E + 7) / 8, 256, 0, stream>>>(srcI, dstI, exv, denom, H, Z, E);
    gat_finalize<<<(n + 7) / 8, 256, 0, stream>>>(Z, H, X, denom, gamma, beta, n);
}
